// MiniMaxM2Attention_43233140801847
// MI455X (gfx1250) — matrix operand order repaired and checked by reading, not compiled, not run on hardware
//
#include <hip/hip_runtime.h>
#include <hip/hip_bf16.h>
#include <stdint.h>

// ---------------------------------------------------------------------------
// MiniMax-M2 attention layer for gfx1250 (CDNA5 / MI455X), wave32, WMMA bf16.
// B=1, S=2048, H=3072, NH=48, NKV=8, HD=128, ROT=64, THETA=1e4, EPS=1e-6.
// ~280 GFLOP vs ~9us of HBM traffic at 23.3 TB/s => compute-bound; all four
// GEMMs + both attention matmuls run on v_wmma_f32_16x16x32_bf16 (fp32 acc).
// Block-uniform tiles are staged into LDS by the Tensor Data Mover; fragments
// are assembled with b128 LDS loads + shufflevector (no scalar packing).
// ---------------------------------------------------------------------------

typedef __bf16 bf16_t;
typedef __attribute__((ext_vector_type(16))) __bf16 bf16x16;
typedef __attribute__((ext_vector_type(8)))  __bf16 bf16x8;
typedef __attribute__((ext_vector_type(8)))  float  f32x8;
typedef __attribute__((ext_vector_type(4)))  float  f32x4;
typedef __attribute__((ext_vector_type(4)))  unsigned int u32x4;
typedef __attribute__((ext_vector_type(8)))  int    i32x8;
typedef __attribute__((ext_vector_type(4)))  int    i32x4;

static_assert(sizeof(bf16_t) == 2, "bf16 must be 2 bytes");

__device__ __forceinline__ bf16x16 ath_cat16(bf16x8 lo, bf16x8 hi) {
  return __builtin_shufflevector(lo, hi, 0, 1, 2, 3, 4, 5, 6, 7,
                                 8, 9, 10, 11, 12, 13, 14, 15);
}

// ---- Tensor Data Mover (probe doc: 5 args on clang-22/ROCm7.2, 6 args on
// ---- clang-23/amdgpu-toolchain) -------------------------------------------
#if defined(__has_builtin)
#if __has_builtin(__builtin_amdgcn_tensor_load_to_lds)
#define ATH_HAS_TDM 1
#endif
#endif
#ifndef ATH_HAS_TDM
#define ATH_HAS_TDM 0
#endif

#if ATH_HAS_TDM
// 2D tile load: tile_h rows of tile_w contiguous bf16 elements, row stride
// `stride` elements, written compacted to LDS (row-major [tile_h][tile_w]).
__device__ __forceinline__ void ath_tdm_load_2d(unsigned lds_byte_addr,
                                                const void* gptr,
                                                unsigned tile_w, unsigned tile_h,
                                                unsigned stride,
                                                unsigned tensor_w, unsigned tensor_h) {
  unsigned long long ga = (unsigned long long)gptr;
  u32x4 g0;
  g0[0] = 1u;                                                   // count=1, user D#
  g0[1] = lds_byte_addr;                                        // lds_addr [63:32]
  g0[2] = (unsigned)(ga & 0xffffffffull);                       // global_addr lo
  g0[3] = (unsigned)((ga >> 32) & 0x1ffffffull) | (2u << 30);   // addr hi | type=2
  i32x8 g1;
  g1[0] = (int)(1u << 16);                                      // data_size=1 (2B bf16)
  g1[1] = (int)((tensor_w & 0xffffu) << 16);                    // tensor_dim0[15:0]
  g1[2] = (int)((tensor_w >> 16) | ((tensor_h & 0xffffu) << 16));
  g1[3] = (int)((tensor_h >> 16) | (tile_w << 16));             // tile_dim0
  g1[4] = (int)(tile_h & 0xffffu);                              // tile_dim1
  g1[5] = (int)stride;                                          // tensor_dim0_stride lo
  g1[6] = 0;
  g1[7] = 0;
  i32x4 z4 = {0, 0, 0, 0};
#if __clang_major__ >= 23
  i32x8 z8 = {0, 0, 0, 0, 0, 0, 0, 0};
  __builtin_amdgcn_tensor_load_to_lds(g0, g1, z4, z4, z8, 0);
#else
  __builtin_amdgcn_tensor_load_to_lds(g0, g1, z4, z4, 0);
#endif
}
#endif

// ---------------------------------------------------------------------------
// fp32 -> bf16 conversion (8 elems/thread; all sizes are multiples of 2048)
// ---------------------------------------------------------------------------
__global__ __launch_bounds__(256) void cvt_f32_bf16_kernel(
    const float* __restrict__ src, bf16_t* __restrict__ dst, long n) {
  const long i8 = ((long)blockIdx.x * 256 + threadIdx.x) * 8;
  if (i8 + 7 < n) {
    f32x4 v0 = *(const f32x4*)(src + i8);
    f32x4 v1 = *(const f32x4*)(src + i8 + 4);
    bf16x8 o;
#pragma unroll
    for (int e = 0; e < 4; ++e) { o[e] = (bf16_t)v0[e]; o[4 + e] = (bf16_t)v1[e]; }
    *(bf16x8*)(dst + i8) = o;
  }
}

// ---------------------------------------------------------------------------
// Tiled bf16 WMMA GEMM: C[MxN] = A[MxK] * B[KxN], A/B bf16 row-major,
// C fp32 or bf16. Block 256 thr = 8 waves; BM=BN=128, BK=32; each wave owns a
// 32x64 tile (2x4 subtiles of 16x16). A tile staged by TDM; B tile staged
// transposed ([N][K]) so B-fragments are contiguous b128 LDS reads.
// ---------------------------------------------------------------------------
template <bool OUT_BF16>
__global__ __launch_bounds__(256) void gemm_bf16_wmma(
    const bf16_t* __restrict__ A, const bf16_t* __restrict__ B,
    void* __restrict__ C, int M, int N, int K) {
  constexpr int TBM = 128, TBN = 128, TBK = 32;
  __shared__ __align__(16) bf16_t As[TBM][TBK];   // 8 KB
  __shared__ __align__(16) bf16_t Bs[TBN][TBK];   // 8 KB, transposed

  const int tid  = threadIdx.x;
  const int lane = tid & 31;
  const int wave = tid >> 5;
  const int bn0  = blockIdx.x * TBN;
  const int bm0  = blockIdx.y * TBM;
  const int wm = wave >> 1, wn = wave & 1;        // 4x2 wave grid
  const int nl = lane & 15, hi = lane >> 4;

  f32x8 acc[2][4] = {};

  for (int k0 = 0; k0 < K; k0 += TBK) {
#if ATH_HAS_TDM
    if (wave == 0) {
      ath_tdm_load_2d((unsigned)(unsigned long long)&As[0][0],
                      (const void*)(A + (long)bm0 * K + k0),
                      (unsigned)TBK, (unsigned)TBM, (unsigned)K,
                      (unsigned)(K - k0), (unsigned)(M - bm0));
    }
#else
    {
      int i = tid;
#pragma unroll
      for (int rep = 0; rep < 2; ++rep, i += 256) {
        const int r = i >> 2, cv = i & 3;
        *(bf16x8*)&As[r][cv * 8] =
            *(const bf16x8*)&A[(long)(bm0 + r) * K + k0 + cv * 8];
      }
    }
#endif
    // stage B transposed: Bs[n][k] = B[k0+k][bn0+n]
    {
      int i = tid;
#pragma unroll
      for (int rep = 0; rep < 2; ++rep, i += 256) {
        const int kk = i >> 4, j = i & 15;
        bf16x8 v = *(const bf16x8*)&B[(long)(k0 + kk) * N + bn0 + j * 8];
#pragma unroll
        for (int e = 0; e < 8; ++e) Bs[j * 8 + e][kk] = v[e];
      }
    }
    if (k0 + TBK < K) {  // pull next K-tile toward L2/L0 (global_prefetch_b8)
      __builtin_prefetch(&A[(long)(bm0 + (tid >> 1)) * K + (k0 + TBK)], 0, 1);
      __builtin_prefetch(&B[(long)(k0 + TBK + (tid >> 4)) * N + bn0 + (tid & 15) * 8], 0, 1);
    }
#if ATH_HAS_TDM
    if (wave == 0) __builtin_amdgcn_s_wait_tensorcnt(0);
#endif
    __syncthreads();

    // A frag (16-bit A 16x32): lanes0-15 K=0..7,16..23; lanes16-31 +8
    bf16x16 afrag[2];
#pragma unroll
    for (int i = 0; i < 2; ++i) {
      const int row = wm * 32 + i * 16 + nl;
      afrag[i] = ath_cat16(*(const bf16x8*)&As[row][hi * 8],
                           *(const bf16x8*)&As[row][16 + hi * 8]);
    }
#pragma unroll
    for (int j = 0; j < 4; ++j) {
      const int col = wn * 64 + j * 16 + nl;
      // B frag (32x16): lanes0-15 K=0..7,16..23; lanes16-31 +8 (matches A)
      const bf16x16 bfrag = ath_cat16(*(const bf16x8*)&Bs[col][hi * 8],
                                      *(const bf16x8*)&Bs[col][16 + hi * 8]);
#pragma unroll
      for (int i = 0; i < 2; ++i)
        acc[i][j] = __builtin_amdgcn_wmma_f32_16x16x32_bf16(
            false, afrag[i], false, bfrag, (short)0, acc[i][j], false, false);
    }
    __syncthreads();
  }

  // C layout: VGPR r -> M = r + hi*8, N = lane&15
#pragma unroll
  for (int i = 0; i < 2; ++i)
#pragma unroll
    for (int j = 0; j < 4; ++j)
#pragma unroll
      for (int r = 0; r < 8; ++r) {
        const long row = bm0 + wm * 32 + i * 16 + hi * 8 + r;
        const long col = bn0 + wn * 64 + j * 16 + nl;
        if (OUT_BF16)
          ((bf16_t*)C)[row * N + col] = (bf16_t)acc[i][j][r];
        else
          ((float*)C)[row * N + col] = acc[i][j][r];
      }
}

// ---------------------------------------------------------------------------
// Fused full-dim RMSNorm (over D) + partial RoPE (first 64 of each 128-dim
// head), in-place on bf16 rows. One block per token; pos = blockIdx.x (B==1).
// inv_freq[j] = theta^(-j/32) = exp(-j * ln(1e4)/32)
// ---------------------------------------------------------------------------
__global__ __launch_bounds__(256) void rmsnorm_rope_kernel(
    bf16_t* __restrict__ X, const float* __restrict__ W, int D) {
  __shared__ float ln[6144];
  __shared__ float red[9];
  const int s = blockIdx.x;
  const int tid = threadIdx.x;
  bf16_t* row = X + (long)s * D;

  float ss = 0.f;
  for (int d = tid; d < D; d += 256) {
    const float v = (float)row[d];
    ln[d] = v;
    ss += v * v;
  }
#pragma unroll
  for (int off = 16; off > 0; off >>= 1) ss += __shfl_down(ss, off, 32);
  if ((tid & 31) == 0) red[tid >> 5] = ss;
  __syncthreads();
  if (tid == 0) {
    float t = 0.f;
#pragma unroll
    for (int i = 0; i < 8; ++i) t += red[i];
    red[8] = rsqrtf(t / (float)D + 1e-6f);
  }
  __syncthreads();
  const float rstd = red[8];

  for (int d = tid; d < D; d += 256) {
    const int hd = d & 127;
    float y = ln[d] * rstd * W[d];
    if (hd < 64) {
      const int j = hd & 31;
      const float ang = (float)s * __expf((float)j * -0.2878231366242557f);
      const float c = __cosf(ang), sn = __sinf(ang);
      const int dp = (hd < 32) ? d + 32 : d - 32;
      const float yp = ln[dp] * rstd * W[dp];
      // hd<32: x1*cos - x2*sin ; 32<=hd<64: x2*cos + x1*sin
      y = (hd < 32) ? (y * c - yp * sn) : (y * c + yp * sn);
    }
    row[d] = (bf16_t)y;
  }
}

// ---------------------------------------------------------------------------
// Flash attention (causal, GQA 48q/8kv), bf16 WMMA, online softmax.
// Block: 64 q-rows x 1 head, 256 threads (8 waves). K streamed in 32-row
// tiles (TDM); V staged transposed ([dim][key]) so P*V B-fragments are
// contiguous b128 LDS reads. LDS ~45 KB.
// ---------------------------------------------------------------------------
__global__ __launch_bounds__(256) void flash_attn_kernel(
    const bf16_t* __restrict__ Q, const bf16_t* __restrict__ Kv,
    const bf16_t* __restrict__ Vv, bf16_t* __restrict__ O) {
  constexpr int DQ = 6144, DKV = 1024, HDIM = 128, GQ = 6;
  __shared__ __align__(16) bf16_t Qs[64][HDIM];   // 16 KB
  __shared__ __align__(16) bf16_t Ks[32][HDIM];   // 8 KB
  __shared__ __align__(16) bf16_t Vt[HDIM][32];   // 8 KB, transposed [dim][key]
  __shared__ __align__(16) float  Ss[64][32];     // 8 KB
  __shared__ __align__(16) bf16_t Ps[64][32];     // 4 KB
  __shared__ float rowm[64], rowl[64], rsc[64];

  const int tid = threadIdx.x, lane = tid & 31, wave = tid >> 5;
  const int qt0 = blockIdx.x * 64;
  const int h   = blockIdx.y;
  const int kvh = h / GQ;
  const int nl = lane & 15, hi = lane >> 4;

  // Q tile (64x128 bf16): TDM when available, else vectorized loads
#if ATH_HAS_TDM
  if (wave == 0) {
    ath_tdm_load_2d((unsigned)(unsigned long long)&Qs[0][0],
                    (const void*)(Q + (long)qt0 * DQ + h * HDIM),
                    (unsigned)HDIM, 64u, (unsigned)DQ,
                    (unsigned)HDIM, (unsigned)(2048 - qt0));
  }
#else
#pragma unroll
  for (int rep = 0; rep < 4; ++rep) {
    const int i = tid + rep * 256;
    const int r = i >> 4, c = (i & 15) * 8;
    *(bf16x8*)&Qs[r][c] = *(const bf16x8*)&Q[(long)(qt0 + r) * DQ + h * HDIM + c];
  }
#endif
  if (tid < 64) { rowm[tid] = -3.0e38f; rowl[tid] = 0.f; }

  const int sm = wave >> 1, sn = wave & 1;   // score subtile per wave (4x2)
  const int om = wave >> 1, on = wave & 1;   // O tile: rows om*16.., cols on*64..
  f32x8 oacc[4] = {};
  const float scale = 0.088388347648318447f; // 1/sqrt(128)

#if ATH_HAS_TDM
  if (wave == 0) __builtin_amdgcn_s_wait_tensorcnt(0);
#endif
  __syncthreads();

  for (int kt0 = 0; kt0 < qt0 + 64; kt0 += 32) {
    // K tile via TDM (compact [32][128]); V tile staged transposed by all thr
#if ATH_HAS_TDM
    if (wave == 0) {
      ath_tdm_load_2d((unsigned)(unsigned long long)&Ks[0][0],
                      (const void*)(Kv + (long)kt0 * DKV + kvh * HDIM),
                      (unsigned)HDIM, 32u, (unsigned)DKV,
                      (unsigned)HDIM, (unsigned)(2048 - kt0));
    }
#else
#pragma unroll
    for (int rep = 0; rep < 2; ++rep) {
      const int i = tid + rep * 256;
      const int r = i >> 4, c = (i & 15) * 8;
      *(bf16x8*)&Ks[r][c] = *(const bf16x8*)&Kv[(long)(kt0 + r) * DKV + kvh * HDIM + c];
    }
#endif
#pragma unroll
    for (int rep = 0; rep < 2; ++rep) {
      const int i = tid + rep * 256;
      const int r = i >> 4, c = (i & 15) * 8;   // r = key row, c = dim base
      bf16x8 v = *(const bf16x8*)&Vv[(long)(kt0 + r) * DKV + kvh * HDIM + c];
#pragma unroll
      for (int e = 0; e < 8; ++e) Vt[c + e][r] = v[e];
    }
#if ATH_HAS_TDM
    if (wave == 0) __builtin_amdgcn_s_wait_tensorcnt(0);
#endif
    __syncthreads();

    // S(16x16 subtile per wave) = Q * K^T over HDIM=128 (4 WMMA k-steps)
    {
      f32x8 s8 = {};
      const int arow = sm * 16 + nl;
      const int brow = sn * 16 + nl;   // B[k][n] = Ks[n][k] -> contiguous reads
#pragma unroll
      for (int kk = 0; kk < HDIM; kk += 32) {
        const bf16x16 a = ath_cat16(*(const bf16x8*)&Qs[arow][kk + hi * 8],
                                    *(const bf16x8*)&Qs[arow][kk + 16 + hi * 8]);
        const bf16x16 b = ath_cat16(*(const bf16x8*)&Ks[brow][kk + hi * 8],
                                    *(const bf16x8*)&Ks[brow][kk + 16 + hi * 8]);
        s8 = __builtin_amdgcn_wmma_f32_16x16x32_bf16(false, a, false, b,
                                                     (short)0, s8, false, false);
      }
#pragma unroll
      for (int r = 0; r < 8; ++r) {
        const int lr = sm * 16 + hi * 8 + r;
        const int lc = sn * 16 + nl;
        float v = s8[r] * scale;
        if (kt0 + lc > qt0 + lr) v = -3.0e38f;   // causal mask
        Ss[lr][lc] = v;
      }
    }
    __syncthreads();

    // online softmax update, one thread per q-row
    if (tid < 64) {
      const float mold = rowm[tid];
      float mnew = mold;
#pragma unroll
      for (int c = 0; c < 32; ++c) mnew = fmaxf(mnew, Ss[tid][c]);
      const float sc = __expf(mold - mnew);
      float lsum = 0.f;
#pragma unroll
      for (int c = 0; c < 32; ++c) {
        const float p = __expf(Ss[tid][c] - mnew);
        lsum += p;
        Ps[tid][c] = (bf16_t)p;
      }
      rowl[tid] = rowl[tid] * sc + lsum;
      rowm[tid] = mnew;
      rsc[tid] = sc;
    }
    __syncthreads();

    // rescale accumulators, then O += P(64x32) * V(32x128)
    {
      const int arow = om * 16 + nl;
      const bf16x16 a = ath_cat16(*(const bf16x8*)&Ps[arow][hi * 8],
                                  *(const bf16x8*)&Ps[arow][16 + hi * 8]);
#pragma unroll
      for (int j = 0; j < 4; ++j) {
#pragma unroll
        for (int r = 0; r < 8; ++r) oacc[j][r] *= rsc[om * 16 + hi * 8 + r];
        const int col = on * 64 + j * 16 + nl;   // output dim for this lane
        const bf16x16 b = ath_cat16(*(const bf16x8*)&Vt[col][hi * 8],
                                    *(const bf16x8*)&Vt[col][16 + hi * 8]);
        oacc[j] = __builtin_amdgcn_wmma_f32_16x16x32_bf16(
            false, a, false, b, (short)0, oacc[j], false, false);
      }
    }
    __syncthreads();
  }

#pragma unroll
  for (int j = 0; j < 4; ++j) {
    const int col0 = on * 64 + j * 16;
#pragma unroll
    for (int r = 0; r < 8; ++r) {
      const int lr = om * 16 + hi * 8 + r;
      const float v = oacc[j][r] / rowl[lr];
      O[(long)(qt0 + lr) * DQ + h * HDIM + col0 + nl] = (bf16_t)v;
    }
  }
}

// ---------------------------------------------------------------------------
// Host orchestration. d_in order: x, Wq, Wk, Wv, Wo, q_norm_weight, k_norm_weight.
// Workspace layout (bytes), total 159,383,552 (~152 MB):
//   xb  @ 0           (2048x3072 bf16)     Wqb @ 12,582,912  (3072x6144 bf16)
//   Wkb @ 50,331,648  (3072x1024 bf16)     Wvb @ 56,623,104
//   Wob @ 62,914,560  (6144x3072 bf16)     qb  @ 100,663,296 (2048x6144 bf16)
//   kb  @ 125,829,120 (2048x1024 bf16)     vb  @ 130,023,424
//   ab  @ 134,217,728 (2048x6144 bf16)
// ---------------------------------------------------------------------------
extern "C" void kernel_launch(void* const* d_in, const int* in_sizes, int n_in,
                              void* d_out, int out_size, void* d_ws, size_t ws_size,
                              hipStream_t stream) {
  (void)in_sizes; (void)n_in; (void)out_size; (void)ws_size;
  const float* x  = (const float*)d_in[0];
  const float* Wq = (const float*)d_in[1];
  const float* Wk = (const float*)d_in[2];
  const float* Wv = (const float*)d_in[3];
  const float* Wo = (const float*)d_in[4];
  const float* qw = (const float*)d_in[5];
  const float* kw = (const float*)d_in[6];

  char* ws = (char*)d_ws;
  bf16_t* xb  = (bf16_t*)(ws + 0);
  bf16_t* Wqb = (bf16_t*)(ws + 12582912);
  bf16_t* Wkb = (bf16_t*)(ws + 50331648);
  bf16_t* Wvb = (bf16_t*)(ws + 56623104);
  bf16_t* Wob = (bf16_t*)(ws + 62914560);
  bf16_t* qb  = (bf16_t*)(ws + 100663296);
  bf16_t* kb  = (bf16_t*)(ws + 125829120);
  bf16_t* vb  = (bf16_t*)(ws + 130023424);
  bf16_t* ab  = (bf16_t*)(ws + 134217728);

  // 1) fp32 -> bf16 (sizes are exact multiples of 2048)
  cvt_f32_bf16_kernel<<<3072, 256, 0, stream>>>(x,  xb,  (long)2048 * 3072);
  cvt_f32_bf16_kernel<<<9216, 256, 0, stream>>>(Wq, Wqb, (long)3072 * 6144);
  cvt_f32_bf16_kernel<<<1536, 256, 0, stream>>>(Wk, Wkb, (long)3072 * 1024);
  cvt_f32_bf16_kernel<<<1536, 256, 0, stream>>>(Wv, Wvb, (long)3072 * 1024);
  cvt_f32_bf16_kernel<<<9216, 256, 0, stream>>>(Wo, Wob, (long)6144 * 3072);

  // 2) QKV projections (bf16 out)
  gemm_bf16_wmma<true><<<dim3(48, 16), 256, 0, stream>>>(xb, Wqb, qb, 2048, 6144, 3072);
  gemm_bf16_wmma<true><<<dim3(8,  16), 256, 0, stream>>>(xb, Wkb, kb, 2048, 1024, 3072);
  gemm_bf16_wmma<true><<<dim3(8,  16), 256, 0, stream>>>(xb, Wvb, vb, 2048, 1024, 3072);

  // 3) full-dim RMSNorm + partial RoPE, in place
  rmsnorm_rope_kernel<<<2048, 256, 0, stream>>>(qb, qw, 6144);
  rmsnorm_rope_kernel<<<2048, 256, 0, stream>>>(kb, kw, 1024);

  // 4) causal GQA flash attention
  flash_attn_kernel<<<dim3(32, 48), 256, 0, stream>>>(qb, kb, vb, ab);

  // 5) output projection (fp32 out)
  gemm_bf16_wmma<false><<<dim3(24, 16), 256, 0, stream>>>(ab, Wob, d_out, 2048, 3072, 6144);
}
